// routing_60739427500533
// MI455X (gfx1250) — matrix operand order repaired and checked by reading, not compiled, not run on hardware
//
#include <hip/hip_runtime.h>
#include <hip/hip_bf16.h>

// ---- sizes ----
#define NN      4096
#define N_IN    64
#define DD      256
#define IN_DIM  64
#define N_OUT   16
#define OUT_DIM 64
#define KCOLS   (N_IN * IN_DIM)          // 4096
#define KEXT    (KCOLS + N_IN)           // 4160 = 130 * 32
#define OCOLS   (N_OUT * OUT_DIM)        // 1024
#define KB2     (KEXT / 32)              // 130

typedef __attribute__((ext_vector_type(16))) __bf16        v16bf;
typedef __attribute__((ext_vector_type(8)))  float         v8f;
typedef __attribute__((ext_vector_type(4)))  float         v4f;
typedef __attribute__((ext_vector_type(4)))  unsigned int  v4u;

union BFrag { v16bf v; unsigned short s[16]; unsigned int u[8]; v4u q[2]; };

static __device__ __forceinline__ unsigned short f32_to_bf16(float f) {
    unsigned int u = __float_as_uint(f);
    unsigned int r = u + 0x7FFFu + ((u >> 16) & 1u);   // round-to-nearest-even
    return (unsigned short)(r >> 16);
}

// fast path: round-half-away, one v_perm_b32 packs two bf16 into a dword
static __device__ __forceinline__ unsigned int pack2_bf16(float a, float b) {
    unsigned int ua = __float_as_uint(a) + 0x8000u;
    unsigned int ub = __float_as_uint(b) + 0x8000u;
    // D[15:0] = ua[31:16], D[31:16] = ub[31:16]
    return __builtin_amdgcn_perm(ub, ua, 0x07060302u);
}
static __device__ __forceinline__ unsigned short fast_bf16(float f) {
    return (unsigned short)((__float_as_uint(f) + 0x8000u) >> 16);
}

static __device__ __forceinline__ float gelu_exact(float v) {
    return 0.5f * v * (1.0f + erff(v * 0.70710678118654752f));
}

static __device__ __forceinline__ void load_a(BFrag& a, const unsigned short* p) {
    a.q[0] = *(const v4u*)(p);
    a.q[1] = *(const v4u*)(p + 16);
}
static __device__ __forceinline__ void load_b(BFrag& b, const unsigned short* p) {
    const v4u* q = (const v4u*)p;
    b.q[0] = q[0]; b.q[1] = q[1];
}
// A-fragment from row-major f32 (two runs of 8 contiguous at +0 and +16)
static __device__ __forceinline__ void make_a(BFrag& a, const float* ap) {
    v4f f0 = *(const v4f*)(ap);
    v4f f1 = *(const v4f*)(ap + 4);
    v4f f2 = *(const v4f*)(ap + 16);
    v4f f3 = *(const v4f*)(ap + 20);
    a.u[0] = pack2_bf16(f0.x, f0.y); a.u[1] = pack2_bf16(f0.z, f0.w);
    a.u[2] = pack2_bf16(f1.x, f1.y); a.u[3] = pack2_bf16(f1.z, f1.w);
    a.u[4] = pack2_bf16(f2.x, f2.y); a.u[5] = pack2_bf16(f2.z, f2.w);
    a.u[6] = pack2_bf16(f3.x, f3.y); a.u[7] = pack2_bf16(f3.z, f3.w);
}

// ---------- prep: Wcap [64][256][64] f32 -> bf16 B-fragments ----------
// frag f = ((i*8 + kb)*4 + nb); lane L, elem e: k = kb*32 + (e&7) + 8*(L>>4) + 16*(e>>3), n = nb*16 + (L&15)
__global__ void prep_wcap(const float* __restrict__ Wcap, unsigned short* __restrict__ WcapShuf) {
    int t = blockIdx.x * 256 + threadIdx.x;            // 65536
    int lane = t & 31, f = t >> 5;
    int nb = f & 3, kb = (f >> 2) & 7, i = f >> 5;
    int hi = lane >> 4, l15 = lane & 15;
    int n = nb * 16 + l15;
    unsigned short* dst = WcapShuf + ((size_t)f << 9) + lane * 16;
#pragma unroll
    for (int e = 0; e < 16; ++e) {
        int k = kb * 32 + (e & 7) + 8 * hi + 16 * (e >> 3);
        dst[e] = f32_to_bf16(Wcap[((size_t)i * DD + k) * IN_DIM + n]);
    }
}

// ---------- prep: [Wvotes ; Bvotes] -> bf16 B-fragments, K extended to 4160 ----------
__global__ void prep_wvotes(const float* __restrict__ Wv, const float* __restrict__ Bv,
                            unsigned short* __restrict__ WvShuf) {
    int t = blockIdx.x * 256 + threadIdx.x;            // 266240
    if (t >= KB2 * 64 * 32) return;
    int lane = t & 31, f = t >> 5;                     // f = kb*64 + nb
    int nb = f & 63, kb = f >> 6;
    int hi = lane >> 4, l15 = lane & 15;
    int col = nb * 16 + l15;
    int j = col >> 6, hh = col & 63;
    unsigned short* dst = WvShuf + ((size_t)f << 9) + lane * 16;
#pragma unroll
    for (int e = 0; e < 16; ++e) {
        int K = kb * 32 + (e & 7) + 8 * hi + 16 * (e >> 3);
        float v;
        if (K < KCOLS) {
            int i = K >> 6, cc = K & 63;
            v = Wv[(((size_t)i * N_OUT + j) * IN_DIM + cc) * OUT_DIM + hh];
        } else {
            int i = K - KCOLS;
            v = Bv[((size_t)i * N_OUT + j) * OUT_DIM + hh];
        }
        dst[e] = f32_to_bf16(v);
    }
}

// ---------- prep: mask -> last 64 K-columns of hmExt ----------
__global__ void fill_mask(const float* __restrict__ mask, unsigned short* __restrict__ hmExt) {
    int t = blockIdx.x * 256 + threadIdx.x;            // 262144
    int n = t >> 6, i = t & 63;
    hmExt[(size_t)n * KEXT + KCOLS + i] = f32_to_bf16(mask[t]);
}

// ---------- GEMM1: h = gelu(x_i @ Wcap_i + Bcap_i) * mask, row-major bf16 out ----------
// one wave = 32 samples x 64 hidden cols for one capsule, K = 256
__global__ void __launch_bounds__(128)
caps_gemm1(const float* __restrict__ x, const float* __restrict__ mask,
           const float* __restrict__ Bcap, const unsigned short* __restrict__ WcapShuf,
           unsigned short* __restrict__ hmExt) {
    __shared__ __align__(16) unsigned short lds[4][32 * 72];
    const int warp = threadIdx.x >> 5, lane = threadIdx.x & 31;
    const int tile = blockIdx.x * 4 + warp;            // 8192 tiles
    const int i  = tile & 63;                          // capsule
    const int n0 = (tile >> 6) << 5;                   // 32-sample tile base
    const int l15 = lane & 15, hi = lane >> 4;

    v8f c[2][4];
#pragma unroll
    for (int m = 0; m < 2; ++m)
#pragma unroll
        for (int nb = 0; nb < 4; ++nb)
#pragma unroll
            for (int e = 0; e < 8; ++e) c[m][nb][e] = 0.0f;

    const float* xr0 = x + ((size_t)(n0 + l15) * N_IN + i) * DD + hi * 8;
    const float* xr1 = xr0 + (size_t)16 * N_IN * DD;
    const unsigned short* wbase = WcapShuf + ((size_t)i << 14) + lane * 16;

#pragma unroll 2
    for (int kb = 0; kb < 8; ++kb) {
        BFrag a[2], b[4];
        make_a(a[0], xr0 + kb * 32);
        make_a(a[1], xr1 + kb * 32);
        const unsigned short* bp = wbase + ((size_t)kb << 11);   // kb*4 frags * 512
#pragma unroll
        for (int nb = 0; nb < 4; ++nb) load_b(b[nb], bp + (nb << 9));
#pragma unroll
        for (int m = 0; m < 2; ++m)
#pragma unroll
            for (int nb = 0; nb < 4; ++nb)
                c[m][nb] = __builtin_amdgcn_wmma_f32_16x16x32_bf16(
                               false, a[m].v, false, b[nb].v, (short)0, c[m][nb], false, false);
    }

    float bias[4];
#pragma unroll
    for (int nb = 0; nb < 4; ++nb) bias[nb] = Bcap[i * IN_DIM + nb * 16 + l15];
    float mrow[2][8];
#pragma unroll
    for (int m = 0; m < 2; ++m)
#pragma unroll
        for (int r = 0; r < 8; ++r)
            mrow[m][r] = mask[(size_t)(n0 + 16 * m + 8 * hi + r) * N_IN + i];

    unsigned short* lw = lds[warp];
#pragma unroll
    for (int m = 0; m < 2; ++m)
#pragma unroll
        for (int nb = 0; nb < 4; ++nb)
#pragma unroll
            for (int r = 0; r < 8; ++r) {
                float v = gelu_exact(c[m][nb][r] + bias[nb]) * mrow[m][r];
                lw[(16 * m + 8 * hi + r) * 72 + nb * 16 + l15] = fast_bf16(v);
            }
    asm volatile("s_wait_dscnt 0x0" ::: "memory");     // per-wave LDS transpose, DS in-order

    // lane L copies row L (64 bf16 = 128 B) to row-major hmExt
    const unsigned short* src = lw + lane * 72;
    unsigned short* dst = hmExt + (size_t)(n0 + lane) * KEXT + i * IN_DIM;
#pragma unroll
    for (int k = 0; k < 8; ++k)
        *(v4u*)(dst + 8 * k) = *(const v4u*)(src + 8 * k);
}

// ---------- GEMM2: out = gelu(hmExt[4096x4160] @ WvExt[4160x1024]) ----------
// one wave = 32 samples x 64 output cols, K = 4160 (130 steps, ping-pong pipelined)
__global__ void __launch_bounds__(128)
votes_gemm2(const unsigned short* __restrict__ hmExt,
            const unsigned short* __restrict__ WvShuf,
            float* __restrict__ out) {
    const int warp = threadIdx.x >> 5, lane = threadIdx.x & 31;
    const int tile = blockIdx.x * 4 + warp;            // 2048 tiles
    const int colt = tile & 15;
    const int n0   = (tile >> 4) << 5;                 // 32 rows
    const int col0 = colt << 6;
    const int l15 = lane & 15, hi = lane >> 4;

    v8f c[2][4];
#pragma unroll
    for (int m = 0; m < 2; ++m)
#pragma unroll
        for (int nb = 0; nb < 4; ++nb)
#pragma unroll
            for (int e = 0; e < 8; ++e) c[m][nb][e] = 0.0f;

    const unsigned short* ar0 = hmExt + (size_t)(n0 + l15) * KEXT + hi * 8;
    const unsigned short* ar1 = ar0 + (size_t)16 * KEXT;
    const unsigned short* bb  = WvShuf + (((size_t)colt * 4) << 9) + lane * 16;

    BFrag a0[2], b0[4], a1[2], b1[4];

#define LD(A, B, kb)                                                   \
    do {                                                               \
        load_a((A)[0], ar0 + (kb) * 32);                               \
        load_a((A)[1], ar1 + (kb) * 32);                               \
        const unsigned short* bp_ = bb + (((size_t)(kb) * 64) << 9);   \
        load_b((B)[0], bp_);        load_b((B)[1], bp_ + 512);         \
        load_b((B)[2], bp_ + 1024); load_b((B)[3], bp_ + 1536);        \
    } while (0)

#define MM(A, B)                                                       \
    do {                                                               \
        _Pragma("unroll")                                              \
        for (int m_ = 0; m_ < 2; ++m_)                                 \
            _Pragma("unroll")                                          \
            for (int nb_ = 0; nb_ < 4; ++nb_)                          \
                c[m_][nb_] = __builtin_amdgcn_wmma_f32_16x16x32_bf16(  \
                    false, (A)[m_].v, false, (B)[nb_].v, (short)0,     \
                    c[m_][nb_], false, false);                         \
    } while (0)

    LD(a0, b0, 0);
#pragma unroll 1
    for (int kb = 0; kb < KB2 - 2; kb += 2) {
        LD(a1, b1, kb + 1);
        MM(a0, b0);
        LD(a0, b0, kb + 2);
        MM(a1, b1);
    }
    LD(a1, b1, KB2 - 1);
    MM(a0, b0);
    MM(a1, b1);
#undef LD
#undef MM

#pragma unroll
    for (int m = 0; m < 2; ++m)
#pragma unroll
        for (int nb = 0; nb < 4; ++nb)
#pragma unroll
            for (int r = 0; r < 8; ++r) {
                out[(size_t)(n0 + 16 * m + 8 * hi + r) * OCOLS + col0 + nb * 16 + l15] =
                    gelu_exact(c[m][nb][r]);
            }
}

// ---------- launch ----------
extern "C" void kernel_launch(void* const* d_in, const int* in_sizes, int n_in,
                              void* d_out, int out_size, void* d_ws, size_t ws_size,
                              hipStream_t stream) {
    const float* x     = (const float*)d_in[0];
    const float* mask  = (const float*)d_in[1];
    const float* Wcap  = (const float*)d_in[2];
    const float* Bcap  = (const float*)d_in[3];
    const float* Wv    = (const float*)d_in[4];
    const float* Bv    = (const float*)d_in[5];
    float* out = (float*)d_out;

    const size_t HM_BYTES   = (size_t)NN * KEXT * 2;                 // 34,078,720
    const size_t WCAP_BYTES = (size_t)N_IN * 8 * 4 * 1024;           //  2,097,152
    unsigned short* hmExt    = (unsigned short*)d_ws;
    unsigned short* WcapShuf = (unsigned short*)((char*)d_ws + HM_BYTES);
    unsigned short* WvShuf   = (unsigned short*)((char*)d_ws + HM_BYTES + WCAP_BYTES);

    prep_wcap  <<<256,  256, 0, stream>>>(Wcap, WcapShuf);
    prep_wvotes<<<1040, 256, 0, stream>>>(Wv, Bv, WvShuf);
    fill_mask  <<<1024, 256, 0, stream>>>(mask, hmExt);
    caps_gemm1 <<<2048, 128, 0, stream>>>(x, mask, Bcap, WcapShuf, hmExt);
    votes_gemm2<<< 512, 128, 0, stream>>>(hmExt, WvShuf, out);
}
